// Hopfield_68092411511384
// MI455X (gfx1250) — matrix operand order repaired and checked by reading, compiled, not run on hardware
//
#include <hip/hip_runtime.h>
#include <hip/hip_bf16.h>

typedef __attribute__((ext_vector_type(16))) _Float16 v16h;
typedef __attribute__((ext_vector_type(8)))  _Float16 v8h;
typedef __attribute__((ext_vector_type(8)))  float    v8f;
typedef __attribute__((ext_vector_type(4)))  float    v4f;

#define B_   8
#define S_   2048
#define D_   256
#define H_   4
#define HD_  64
#define NROWS (B_*S_)   // 16384

// workspace layout (bytes), 8MB regions
#define OFF_SN  ((size_t)0)          // f16 [16384][256] stored-LN
#define OFF_QN  ((size_t)8  << 20)   // f16 [16384][256] state-LN
#define OFF_VN  ((size_t)16 << 20)   // f16 [16384][256] proj-LN
#define OFF_QH  ((size_t)24 << 20)   // f16 [B,H,S,HD]  q (scaled)
#define OFF_KH  ((size_t)32 << 20)   // f16 [B,H,S,HD]  k
#define OFF_VT  ((size_t)40 << 20)   // f16 [B,H,HD,S]  v transposed
#define OFF_AO  ((size_t)48 << 20)   // f16 [B,S,D]     attention out
#define OFF_W16 ((size_t)56 << 20)   // f16 Wq,Wk,Wv,Wo (4 x 65536)

// ---------------- WMMA helpers ----------------
__device__ __forceinline__ v8f wmma32(v16h a, v16h b, v8f c) {
  return __builtin_amdgcn_wmma_f32_16x16x32_f16(false, a, false, b, (short)0, c,
                                                false, false);
}

// A operand: 16x32 f16 tile from row-major [.. , ld] source.
// lane L (L<16): row=L, k = {0..15}; lane L>=16: row=L-16, k = {16..31}
__device__ __forceinline__ v16h fragA(const _Float16* base, int ld, int row0,
                                      int k0, int lane) {
  int r = lane & 15, hh = lane >> 4;
  const _Float16* p = base + (size_t)(row0 + r) * ld + k0 + hh * 16;
  v8h lo = *(const v8h*)p;
  v8h hi = *(const v8h*)(p + 8);
  v16h f;
#pragma unroll
  for (int i = 0; i < 8; ++i) { f[i] = lo[i]; f[i + 8] = hi[i]; }
  return f;
}

// B operand for C = A * T^T where T ([n][k]) is row-major with leading dim ld:
// lane L: n = L&15, holds k = (L>>4)*16 .. +15 contiguous (one 32B load).
__device__ __forceinline__ v16h fragB(const _Float16* baseT, int ld, int n0,
                                      int k0, int lane) {
  int n = lane & 15, hh = lane >> 4;
  return *(const v16h*)(baseT + (size_t)(n0 + n) * ld + k0 + hh * 16);
}

// ---------------- kernel 1: f32 -> f16 weight conversion ----------------
__global__ __launch_bounds__(256) void cvt_w_kernel(const float* wq, const float* wk,
                                                    const float* wv, const float* wo,
                                                    _Float16* w16) {
  int i = blockIdx.x * 256 + threadIdx.x;       // 0 .. 262143
  int m = i >> 16, e = i & 65535;
  const float* src = (m == 0) ? wq : (m == 1) ? wk : (m == 2) ? wv : wo;
  w16[i] = (_Float16)src[e];
}

// ---------------- kernel 2: fused triple LayerNorm ----------------
__global__ __launch_bounds__(256) void ln_kernel(const float* __restrict__ x,
    const float* gs, const float* bs, const float* gq, const float* bq,
    const float* gv, const float* bv,
    _Float16* sn, _Float16* qn, _Float16* vn) {
  int lane = threadIdx.x & 31;
  int row  = blockIdx.x * 8 + (threadIdx.x >> 5);
  const float* xr = x + (size_t)row * D_ + lane * 8;
  v4f a  = *(const v4f*)xr;
  v4f b4 = *(const v4f*)(xr + 4);
  float xv[8];
#pragma unroll
  for (int j = 0; j < 4; ++j) { xv[j] = a[j]; xv[4 + j] = b4[j]; }
  float s = 0.f, ss = 0.f;
#pragma unroll
  for (int j = 0; j < 8; ++j) { s += xv[j]; ss += xv[j] * xv[j]; }
#pragma unroll
  for (int m = 1; m < 32; m <<= 1) {
    s  += __shfl_xor(s, m, 32);
    ss += __shfl_xor(ss, m, 32);
  }
  float mean = s * (1.f / 256.f);
  float var  = ss * (1.f / 256.f) - mean * mean;
  float rstd = rsqrtf(var + 1e-5f);
  int c0 = lane * 8;
  v8h o1, o2, o3;
#pragma unroll
  for (int j = 0; j < 8; ++j) {
    float nx = (xv[j] - mean) * rstd;
    o1[j] = (_Float16)(nx * gs[c0 + j] + bs[c0 + j]);
    o2[j] = (_Float16)(nx * gq[c0 + j] + bq[c0 + j]);
    o3[j] = (_Float16)(nx * gv[c0 + j] + bv[c0 + j]);
  }
  size_t o = (size_t)row * D_ + c0;
  *(v8h*)(sn + o) = o1;
  *(v8h*)(qn + o) = o2;
  *(v8h*)(vn + o) = o3;
}

// ---------------- kernel 3: QKV projections, 16x64 tile per wave ----------------
// block = 4 waves; wave w computes cols [w*64, w*64+64) == head w for its 16-row tile.
__global__ __launch_bounds__(128) void qkv_kernel(
    const _Float16* __restrict__ sn, const _Float16* __restrict__ qn,
    const _Float16* __restrict__ vn, const _Float16* __restrict__ w16,
    const float* bq, const float* bk, const float* bv,
    _Float16* __restrict__ qh, _Float16* __restrict__ kh,
    _Float16* __restrict__ vt) {
  int lane = threadIdx.x & 31;
  int w    = threadIdx.x >> 5;          // head / 64-col group
  int rt   = blockIdx.x;                // 16-row tile of B*S
  int mat  = blockIdx.y;                // 0=q, 1=k, 2=v
  const _Float16* A = (mat == 0) ? qn : (mat == 1) ? sn : vn;
  const _Float16* W = w16 + (size_t)mat * 65536 + (size_t)w * 64 * D_;
  const float* bias = (mat == 0) ? bq : (mat == 1) ? bk : bv;
  v8f acc0 = {}, acc1 = {}, acc2 = {}, acc3 = {};
#pragma unroll
  for (int k0 = 0; k0 < D_; k0 += 32) {
    v16h a = fragA(A, D_, rt * 16, k0, lane);   // reused across 4 N-tiles
    acc0 = wmma32(a, fragB(W, D_, 0,  k0, lane), acc0);
    acc1 = wmma32(a, fragB(W, D_, 16, k0, lane), acc1);
    acc2 = wmma32(a, fragB(W, D_, 32, k0, lane), acc2);
    acc3 = wmma32(a, fragB(W, D_, 48, k0, lane), acc3);
  }
  int hh = lane >> 4, n = lane & 15;
  int head = w;
  float bc[4];
#pragma unroll
  for (int t = 0; t < 4; ++t) bc[t] = bias[head * 64 + t * 16 + n];
#pragma unroll
  for (int r = 0; r < 8; ++r) {
    int row = rt * 16 + hh * 8 + r;
    int bb = row >> 11, sidx = row & (S_ - 1);
    float v0 = acc0[r] + bc[0], v1 = acc1[r] + bc[1];
    float v2 = acc2[r] + bc[2], v3 = acc3[r] + bc[3];
    size_t hb = (size_t)(bb * H_ + head);
    if (mat == 0) {
      _Float16* p = qh + (hb * S_ + sidx) * HD_ + n;
      p[0] = (_Float16)(v0 * 0.125f); p[16] = (_Float16)(v1 * 0.125f);
      p[32] = (_Float16)(v2 * 0.125f); p[48] = (_Float16)(v3 * 0.125f);
    } else if (mat == 1) {
      _Float16* p = kh + (hb * S_ + sidx) * HD_ + n;
      p[0] = (_Float16)v0; p[16] = (_Float16)v1;
      p[32] = (_Float16)v2; p[48] = (_Float16)v3;
    } else {
      _Float16* p = vt + hb * HD_ * S_ + (size_t)n * S_ + sidx;
      p[0]            = (_Float16)v0;
      p[16 * S_]      = (_Float16)v1;
      p[32 * S_]      = (_Float16)v2;
      p[48 * S_]      = (_Float16)v3;
    }
  }
}

// ---------------- kernel 4: flash attention, one 16-query tile per wave ----------------
#define FA_WAVES 8
__global__ __launch_bounds__(32 * FA_WAVES) void attn_kernel(
    const _Float16* __restrict__ qh, const _Float16* __restrict__ kh,
    const _Float16* __restrict__ vt, _Float16* __restrict__ ao) {
  __shared__ _Float16 plds[FA_WAVES][16][40];  // P tile staging (padded rows)
  int lane = threadIdx.x & 31;
  int w    = threadIdx.x >> 5;
  int t    = blockIdx.x * FA_WAVES + w;  // 0..4095
  int bh = t >> 7;        // (b*H + h)
  int qt = t & 127;       // query tile within S
  int bb = bh >> 2, head = bh & 3;
  const _Float16* qb = qh + (size_t)bh * S_ * HD_;
  const _Float16* kb = kh + (size_t)bh * S_ * HD_;
  const _Float16* vb = vt + (size_t)bh * HD_ * S_;

  v16h qa0 = fragA(qb, HD_, qt * 16, 0, lane);
  v16h qa1 = fragA(qb, HD_, qt * 16, 32, lane);
  v8f o0 = {}, o1 = {}, o2 = {}, o3 = {};
  float m[8], l[8];
#pragma unroll
  for (int r = 0; r < 8; ++r) { m[r] = -1e30f; l[r] = 0.f; }
  int hh = lane >> 4;

  for (int kbl = 0; kbl < S_; kbl += 32) {
    if (kbl + 32 < S_) {
      __builtin_prefetch(kb + (size_t)(kbl + 32) * HD_, 0, 1);
      __builtin_prefetch(vb + kbl + 32, 0, 1);
    }
    // scores: 16 queries x 32 keys, K-dim = HD = 64
    v8f s0 = {}, s1 = {};
    s0 = wmma32(qa0, fragB(kb, HD_, kbl,      0,  lane), s0);
    s0 = wmma32(qa1, fragB(kb, HD_, kbl,      32, lane), s0);
    s1 = wmma32(qa0, fragB(kb, HD_, kbl + 16, 0,  lane), s1);
    s1 = wmma32(qa1, fragB(kb, HD_, kbl + 16, 32, lane), s1);

    // online softmax: rows are striped across the 16 lanes of each half-wave
#pragma unroll
    for (int r = 0; r < 8; ++r) {
      float v = fmaxf(s0[r], s1[r]);
      v = fmaxf(v, __shfl_xor(v, 1, 32));
      v = fmaxf(v, __shfl_xor(v, 2, 32));
      v = fmaxf(v, __shfl_xor(v, 4, 32));
      v = fmaxf(v, __shfl_xor(v, 8, 32));
      float mn = fmaxf(m[r], v);
      float al = __expf(m[r] - mn);
      m[r] = mn;
      float p0 = __expf(s0[r] - mn);
      float p1 = __expf(s1[r] - mn);
      float rs = p0 + p1;
      rs += __shfl_xor(rs, 1, 32);
      rs += __shfl_xor(rs, 2, 32);
      rs += __shfl_xor(rs, 4, 32);
      rs += __shfl_xor(rs, 8, 32);
      l[r] = l[r] * al + rs;
      o0[r] *= al; o1[r] *= al; o2[r] *= al; o3[r] *= al;
      int row = hh * 8 + r, col = lane & 15;
      plds[w][row][col]      = (_Float16)p0;
      plds[w][row][col + 16] = (_Float16)p1;
    }
    asm volatile("s_wait_dscnt 0x0" ::: "memory");  // LDS store->load within wave

    // out += P (16x32) @ V (32 keys x 64 d); V^T rows come from vt [HD][S]
    v16h pf = fragA(&plds[w][0][0], 40, 0, 0, lane);
    o0 = wmma32(pf, fragB(vb, S_, 0,  kbl, lane), o0);
    o1 = wmma32(pf, fragB(vb, S_, 16, kbl, lane), o1);
    o2 = wmma32(pf, fragB(vb, S_, 32, kbl, lane), o2);
    o3 = wmma32(pf, fragB(vb, S_, 48, kbl, lane), o3);
  }

#pragma unroll
  for (int r = 0; r < 8; ++r) {
    float rl = 1.0f / l[r];
    int row = bb * S_ + qt * 16 + hh * 8 + r;
    int n = lane & 15;
    size_t base = (size_t)row * D_ + head * HD_ + n;
    ao[base + 0]  = (_Float16)(o0[r] * rl);
    ao[base + 16] = (_Float16)(o1[r] * rl);
    ao[base + 32] = (_Float16)(o2[r] * rl);
    ao[base + 48] = (_Float16)(o3[r] * rl);
  }
}

// ---------------- kernel 5: output projection, 16x64 tile per wave (f32 out) ----------------
__global__ __launch_bounds__(128) void oproj_kernel(const _Float16* __restrict__ ao,
                                                    const _Float16* __restrict__ wo16,
                                                    const float* bo,
                                                    float* __restrict__ out) {
  int lane = threadIdx.x & 31;
  int w    = threadIdx.x >> 5;
  int rt   = blockIdx.x;
  const _Float16* W = wo16 + (size_t)w * 64 * D_;
  v8f acc0 = {}, acc1 = {}, acc2 = {}, acc3 = {};
#pragma unroll
  for (int k0 = 0; k0 < D_; k0 += 32) {
    v16h a = fragA(ao, D_, rt * 16, k0, lane);
    acc0 = wmma32(a, fragB(W, D_, 0,  k0, lane), acc0);
    acc1 = wmma32(a, fragB(W, D_, 16, k0, lane), acc1);
    acc2 = wmma32(a, fragB(W, D_, 32, k0, lane), acc2);
    acc3 = wmma32(a, fragB(W, D_, 48, k0, lane), acc3);
  }
  int hh = lane >> 4, n = lane & 15;
  int c0 = w * 64 + n;
#pragma unroll
  for (int r = 0; r < 8; ++r) {
    float* p = out + (size_t)(rt * 16 + hh * 8 + r) * D_ + c0;
    p[0]  = acc0[r] + bo[c0];
    p[16] = acc1[r] + bo[c0 + 16];
    p[32] = acc2[r] + bo[c0 + 32];
    p[48] = acc3[r] + bo[c0 + 48];
  }
}

// ---------------- launch ----------------
extern "C" void kernel_launch(void* const* d_in, const int* in_sizes, int n_in,
                              void* d_out, int out_size, void* d_ws, size_t ws_size,
                              hipStream_t stream) {
  (void)in_sizes; (void)n_in; (void)out_size; (void)ws_size;
  const float* x    = (const float*)d_in[0];
  const float* lnsg = (const float*)d_in[1];
  const float* lnsb = (const float*)d_in[2];
  const float* lnqg = (const float*)d_in[3];
  const float* lnqb = (const float*)d_in[4];
  const float* lnvg = (const float*)d_in[5];
  const float* lnvb = (const float*)d_in[6];
  const float* Wq = (const float*)d_in[7];
  const float* bq = (const float*)d_in[8];
  const float* Wk = (const float*)d_in[9];
  const float* bk = (const float*)d_in[10];
  const float* Wv = (const float*)d_in[11];
  const float* bv = (const float*)d_in[12];
  const float* Wo = (const float*)d_in[13];
  const float* bo = (const float*)d_in[14];
  float* out = (float*)d_out;

  char* ws = (char*)d_ws;
  _Float16* sn  = (_Float16*)(ws + OFF_SN);
  _Float16* qn  = (_Float16*)(ws + OFF_QN);
  _Float16* vn  = (_Float16*)(ws + OFF_VN);
  _Float16* qhp = (_Float16*)(ws + OFF_QH);
  _Float16* khp = (_Float16*)(ws + OFF_KH);
  _Float16* vtp = (_Float16*)(ws + OFF_VT);
  _Float16* aop = (_Float16*)(ws + OFF_AO);
  _Float16* w16 = (_Float16*)(ws + OFF_W16);

  cvt_w_kernel<<<1024, 256, 0, stream>>>(Wq, Wk, Wv, Wo, w16);
  ln_kernel<<<NROWS / 8, 256, 0, stream>>>(x, lnsg, lnsb, lnqg, lnqb, lnvg, lnvb,
                                           sn, qn, vn);
  qkv_kernel<<<dim3(NROWS / 16, 3), 128, 0, stream>>>(
      sn, qn, vn, w16, bq, bk, bv, qhp, khp, vtp);
  attn_kernel<<<(B_ * H_ * (S_ / 16)) / FA_WAVES, 32 * FA_WAVES, 0, stream>>>(
      qhp, khp, vtp, aop);
  oproj_kernel<<<NROWS / 16, 128, 0, stream>>>(aop, w16 + 3 * 65536, bo, out);
}
